// QKVAttention_57226144252723
// MI455X (gfx1250) — compile-verified, matrix operand order checked by reading, not run on hardware
//
#include <hip/hip_runtime.h>
#include <hip/hip_bf16.h>

// ---------------------------------------------------------------------------
// Problem constants: B=2, T=2048, C=1024, H=16, DK=DV=64
// ---------------------------------------------------------------------------
#define TB    2
#define TT    2048
#define TC    1024
#define TH    16
#define TD    64
#define MROWS (TB * TT)          // 4096

typedef __bf16 bf16;
typedef __attribute__((ext_vector_type(16))) __bf16 v16bf;
typedef __attribute__((ext_vector_type(8)))  __bf16 v8bf;
typedef __attribute__((ext_vector_type(8)))  float  v8f;

// ---------------------------------------------------------------------------
// CDNA5 async copy: global -> LDS, 16B per lane, tracked by ASYNCcnt.
// LDS byte offset = low 32 bits of the generic pointer (flat LDS address is
// {aperture_hi32, lds_offset}).
// ---------------------------------------------------------------------------
static __device__ inline void async_b128(const bf16* g, bf16* l) {
  unsigned loff = (unsigned)(size_t)l;
  unsigned long long ga = (unsigned long long)(size_t)g;
  asm volatile("global_load_async_to_lds_b128 %0, %1, off"
               :: "v"(loff), "v"(ga) : "memory");
}
static __device__ inline void wait_async() {
  asm volatile("s_wait_asynccnt 0" ::: "memory");
}

// ---------------------------------------------------------------------------
// WMMA fragment load: 16x32 bf16 tile per CDNA5 16-bit A-matrix layout
// (lanes 0-15: K=0..7 -> elts 0-7, K=16..23 -> elts 8-15;
//  lanes 16-31: K=8..15 / K=24..31). B (32x16) mirrors over columns.
// Works for global or LDS pointers (generic); all chunks 16B aligned.
// ---------------------------------------------------------------------------
static __device__ inline v16bf load_frag16x32(const bf16* __restrict__ src,
                                              int ld, int row_base, int k_base,
                                              int lane) {
  const int r    = lane & 15;
  const int ksub = (lane >> 4) << 3;   // 0 or 8
  const bf16* p  = src + (size_t)(row_base + r) * (size_t)ld + (size_t)(k_base + ksub);
  v8bf lo = *(const v8bf*)(p);
  v8bf hi = *(const v8bf*)(p + 16);
  v16bf f;
#pragma unroll
  for (int i = 0; i < 8; ++i) { f[i] = lo[i]; f[i + 8] = hi[i]; }
  return f;
}

static __device__ inline v8f wmma_bf16(v16bf a, v16bf b, v8f c) {
  return __builtin_amdgcn_wmma_f32_16x16x32_bf16(false, a, false, b,
                                                 (short)0, c, false, false);
}

// ---------------------------------------------------------------------------
// fp32 -> bf16 conversion kernels
// ---------------------------------------------------------------------------
__global__ void cvt_f32_bf16(const float* __restrict__ src, bf16* __restrict__ dst, int n) {
  for (int i = blockIdx.x * blockDim.x + threadIdx.x; i < n; i += gridDim.x * blockDim.x)
    dst[i] = (bf16)src[i];
}

// w_o [1024][1024] (K-major) -> w_o^T (N-major, K contiguous), bf16
__global__ void transpose_wo_bf16(const float* __restrict__ src, bf16* __restrict__ dst) {
  int i = blockIdx.x * blockDim.x + threadIdx.x;
  int j = i >> 10, c = i & 1023;
  dst[(size_t)c * 1024 + j] = (bf16)src[i];
}

// ---------------------------------------------------------------------------
// Block-tiled GEMM staging: A tile 64x32 (256 x 16B), B tile 128x32 (512 x 16B)
// ---------------------------------------------------------------------------
static __device__ inline void stage_gemm(const bf16* __restrict__ A, int arow0,
                                         const bf16* __restrict__ Bm, int brow0,
                                         int k0, bf16* sA, bf16* sB, int tid) {
  const int row = tid >> 2, ch = (tid & 3) << 3;      // ch in elements
  async_b128(A  + (size_t)(arow0 + row) * TC + k0 + ch, sA + row * 32 + ch);
  async_b128(Bm + (size_t)(brow0 + row) * TC + k0 + ch, sB + row * 32 + ch);
  async_b128(Bm + (size_t)(brow0 + row + 64) * TC + k0 + ch, sB + (row + 64) * 32 + ch);
}

// ---------------------------------------------------------------------------
// QKV projection: [4096 x 1024] x W^T, N = 3*1024.  Block = 64(M) x 128(N),
// 8 waves as 2(M) x 4(N), each wave 32x32 (4 accumulators).  Double-buffered
// LDS tiles filled with async copies.  q gets 1/sqrt(DK)=0.125 folded in.
// q,k stored [B*H][T][DK]; v stored transposed [B*H][DV][T].
// ---------------------------------------------------------------------------
__global__ __launch_bounds__(256) void qkv_gemm(const bf16* __restrict__ xb,
                                                const bf16* __restrict__ wqb,
                                                const bf16* __restrict__ wkb,
                                                const bf16* __restrict__ wvb,
                                                bf16* __restrict__ qb,
                                                bf16* __restrict__ kb,
                                                bf16* __restrict__ vt) {
  __shared__ __align__(16) bf16 sA[2][64 * 32];
  __shared__ __align__(16) bf16 sB[2][128 * 32];
  const int tid = threadIdx.x, lane = tid & 31, wid = tid >> 5;
  const int mt = blockIdx.x / 24, nt = blockIdx.x % 24;   // 64 x 24 blocks
  const int mblock = mt << 6;
  const int nAll   = nt << 7;
  const int wsel   = nAll >> 10;            // 0:q 1:k 2:v (uniform per block)
  const int nloc0  = nAll & 1023;
  const bf16* wb = (wsel == 0) ? wqb : (wsel == 1) ? wkb : wvb;
  const int wm = wid & 1, wn = wid >> 1;

  v8f acc[2][2] = {};
  stage_gemm(xb, mblock, wb, nloc0, 0, sA[0], sB[0], tid);
  wait_async();
  __syncthreads();

#pragma unroll 2
  for (int i = 0; i < 32; ++i) {
    const int cur = i & 1;
    if (i + 1 < 32)
      stage_gemm(xb, mblock, wb, nloc0, (i + 1) << 5, sA[cur ^ 1], sB[cur ^ 1], tid);
    v16bf a0 = load_frag16x32(sA[cur], 32, wm * 32,      0, lane);
    v16bf a1 = load_frag16x32(sA[cur], 32, wm * 32 + 16, 0, lane);
    v16bf b0 = load_frag16x32(sB[cur], 32, wn * 32,      0, lane);
    v16bf b1 = load_frag16x32(sB[cur], 32, wn * 32 + 16, 0, lane);
    acc[0][0] = wmma_bf16(a0, b0, acc[0][0]);
    acc[0][1] = wmma_bf16(a0, b1, acc[0][1]);
    acc[1][0] = wmma_bf16(a1, b0, acc[1][0]);
    acc[1][1] = wmma_bf16(a1, b1, acc[1][1]);
    wait_async();
    __syncthreads();
  }

  // Epilogue: f32 D layout -> scattered bf16 stores into q/k/v^T.
  const int col  = lane & 15;
  const int row0 = (lane >> 4) << 3;
#pragma unroll
  for (int mf = 0; mf < 2; ++mf) {
#pragma unroll
    for (int nf = 0; nf < 2; ++nf) {
      const int mb = mblock + wm * 32 + mf * 16;
      const int nb = nloc0 + wn * 32 + nf * 16;
      const int n = nb + col, h = n >> 6, d = n & 63;
#pragma unroll
      for (int r = 0; r < 8; ++r) {
        int m = mb + row0 + r;
        int bi = m >> 11, t = m & (TT - 1);
        size_t bh = (size_t)(bi * TH + h);
        float v = acc[mf][nf][r];
        if (wsel == 0)      qb[(bh * TT + t) * TD + d] = (bf16)(v * 0.125f);
        else if (wsel == 1) kb[(bh * TT + t) * TD + d] = (bf16)v;
        else                vt[(bh * TD + d) * TT + t] = (bf16)v;
      }
    }
  }
}

// ---------------------------------------------------------------------------
// Attention K/V staging: K tile 32x64 (256 x 16B), V tile 64x32 (256 x 16B)
// ---------------------------------------------------------------------------
static __device__ inline void stage_attn(const bf16* __restrict__ kp,
                                         const bf16* __restrict__ vp,
                                         int jb, bf16* sk, bf16* sv, int tid) {
  { const int row = tid >> 3, ch = (tid & 7) << 3;
    async_b128(kp + (size_t)(jb + row) * TD + ch, sk + row * 64 + ch); }
  { const int row = tid >> 2, ch = (tid & 3) << 3;
    async_b128(vp + (size_t)row * TT + jb + ch, sv + row * 32 + ch); }
}

// ---------------------------------------------------------------------------
// Flash-style causal attention.  One block = 128 consecutive queries of one
// (b,h); 8 waves each own 16 queries.  K/V tiles double-buffered in LDS via
// async copies, shared across waves.  P goes D-layout -> A-layout through a
// per-wave LDS scratch.
// ---------------------------------------------------------------------------
__global__ __launch_bounds__(256) void attn_fused(const bf16* __restrict__ qb,
                                                  const bf16* __restrict__ kbm,
                                                  const bf16* __restrict__ vt,
                                                  bf16* __restrict__ yb) {
  __shared__ __align__(16) bf16 sK[2][32 * 64];
  __shared__ __align__(16) bf16 sV[2][64 * 32];
  __shared__ __align__(16) bf16 sP[8][512];
  const int tid = threadIdx.x, lane = tid & 31, wid = tid >> 5;
  bf16* pw = sP[wid];

  const int bh        = blockIdx.x >> 4;    // 32 (b,h) x 16 query-blocks
  const int qblk      = blockIdx.x & 15;
  const int qbase_blk = qblk << 7;
  const int qbase     = qbase_blk + (wid << 4);

  const bf16* qp = qb  + (size_t)bh * TT * TD;
  const bf16* kp = kbm + (size_t)bh * TT * TD;
  const bf16* vp = vt  + (size_t)bh * TD * TT;

  const v16bf aq0 = load_frag16x32(qp, TD, qbase, 0, lane);
  const v16bf aq1 = load_frag16x32(qp, TD, qbase, 32, lane);

  const int col  = lane & 15;
  const int row0 = (lane >> 4) << 3;

  float mi[8], li[8];
#pragma unroll
  for (int r = 0; r < 8; ++r) { mi[r] = -__builtin_inff(); li[r] = 0.0f; }
  v8f o0 = {}, o1 = {}, o2 = {}, o3 = {};

  const int niter = (qbase_blk + 128) >> 5;   // block-uniform
  stage_attn(kp, vp, 0, sK[0], sV[0], tid);
  wait_async();
  __syncthreads();

#pragma unroll 2
  for (int i = 0; i < niter; ++i) {
    const int jb = i << 5, cur = i & 1;
    if (i + 1 < niter)
      stage_attn(kp, vp, jb + 32, sK[cur ^ 1], sV[cur ^ 1], tid);

    if (jb <= qbase + 15) {     // wave-uniform causal skip
      // ---- S = (q*0.125) . k^T  (16 queries x 32 keys) ----
      v16bf k00 = load_frag16x32(sK[cur], TD, 0,  0,  lane);
      v16bf k01 = load_frag16x32(sK[cur], TD, 0,  32, lane);
      v16bf k10 = load_frag16x32(sK[cur], TD, 16, 0,  lane);
      v16bf k11 = load_frag16x32(sK[cur], TD, 16, 32, lane);
      v8f s0 = {}, s1 = {};
      s0 = wmma_bf16(aq0, k00, s0);
      s0 = wmma_bf16(aq1, k01, s0);
      s1 = wmma_bf16(aq0, k10, s1);
      s1 = wmma_bf16(aq1, k11, s1);

      if (jb + 31 > qbase) {    // diagonal tile: per-element causal mask
#pragma unroll
        for (int r = 0; r < 8; ++r) {
          int qg = qbase + row0 + r;
          s0[r] = (jb + col      > qg) ? -__builtin_inff() : s0[r];
          s1[r] = (jb + 16 + col > qg) ? -__builtin_inff() : s1[r];
        }
      }

      // ---- online softmax (row stats over 16-lane halves) ----
      float p0[8], p1[8];
#pragma unroll
      for (int r = 0; r < 8; ++r) {
        float rm = fmaxf(s0[r], s1[r]);
#pragma unroll
        for (int off = 8; off > 0; off >>= 1) rm = fmaxf(rm, __shfl_xor(rm, off, 32));
        float mn = fmaxf(mi[r], rm);
        float al = __expf(mi[r] - mn);
        float e0 = __expf(s0[r] - mn);
        float e1 = __expf(s1[r] - mn);
        float rs = e0 + e1;
#pragma unroll
        for (int off = 8; off > 0; off >>= 1) rs += __shfl_xor(rs, off, 32);
        li[r] = li[r] * al + rs;
        mi[r] = mn;
        o0[r] *= al; o1[r] *= al; o2[r] *= al; o3[r] *= al;
        p0[r] = e0; p1[r] = e1;
      }

      // ---- D-layout P -> per-wave LDS scratch -> A-fragment ----
#pragma unroll
      for (int r = 0; r < 8; ++r) {
        int m = row0 + r;
        pw[m * 32 + col]      = (bf16)p0[r];
        pw[m * 32 + col + 16] = (bf16)p1[r];
      }
      v16bf pa = load_frag16x32(pw, 32, 0, 0, lane);

      // ---- O += P . V ----
      v16bf b0 = load_frag16x32(sV[cur], 32, 0,  0, lane);
      v16bf b1 = load_frag16x32(sV[cur], 32, 16, 0, lane);
      v16bf b2 = load_frag16x32(sV[cur], 32, 32, 0, lane);
      v16bf b3 = load_frag16x32(sV[cur], 32, 48, 0, lane);
      o0 = wmma_bf16(pa, b0, o0);
      o1 = wmma_bf16(pa, b1, o1);
      o2 = wmma_bf16(pa, b2, o2);
      o3 = wmma_bf16(pa, b3, o3);
    }

    wait_async();
    __syncthreads();
  }

  // ---- normalize and store y[b][t][h*64+dv] as bf16 ----
  const int bi = bh >> 4, h = bh & 15;
#pragma unroll
  for (int r = 0; r < 8; ++r) {
    float inv = 1.0f / li[r];
    int t = qbase + row0 + r;
    size_t o = ((size_t)bi * TT + t) * (TH * TD) + h * TD + col;
    yb[o]      = (bf16)(o0[r] * inv);
    yb[o + 16] = (bf16)(o1[r] * inv);
    yb[o + 32] = (bf16)(o2[r] * inv);
    yb[o + 48] = (bf16)(o3[r] * inv);
  }
}

// ---------------------------------------------------------------------------
// Output projection: out[4096][1024] = y x w_o, fp32 out.  Same block tiling.
// ---------------------------------------------------------------------------
__global__ __launch_bounds__(256) void out_gemm(const bf16* __restrict__ yb,
                                                const bf16* __restrict__ wot,
                                                float* __restrict__ out) {
  __shared__ __align__(16) bf16 sA[2][64 * 32];
  __shared__ __align__(16) bf16 sB[2][128 * 32];
  const int tid = threadIdx.x, lane = tid & 31, wid = tid >> 5;
  const int mt = blockIdx.x >> 3, nt = blockIdx.x & 7;    // 64 x 8 blocks
  const int mblock = mt << 6, nblock = nt << 7;
  const int wm = wid & 1, wn = wid >> 1;

  v8f acc[2][2] = {};
  stage_gemm(yb, mblock, wot, nblock, 0, sA[0], sB[0], tid);
  wait_async();
  __syncthreads();

#pragma unroll 2
  for (int i = 0; i < 32; ++i) {
    const int cur = i & 1;
    if (i + 1 < 32)
      stage_gemm(yb, mblock, wot, nblock, (i + 1) << 5, sA[cur ^ 1], sB[cur ^ 1], tid);
    v16bf a0 = load_frag16x32(sA[cur], 32, wm * 32,      0, lane);
    v16bf a1 = load_frag16x32(sA[cur], 32, wm * 32 + 16, 0, lane);
    v16bf b0 = load_frag16x32(sB[cur], 32, wn * 32,      0, lane);
    v16bf b1 = load_frag16x32(sB[cur], 32, wn * 32 + 16, 0, lane);
    acc[0][0] = wmma_bf16(a0, b0, acc[0][0]);
    acc[0][1] = wmma_bf16(a0, b1, acc[0][1]);
    acc[1][0] = wmma_bf16(a1, b0, acc[1][0]);
    acc[1][1] = wmma_bf16(a1, b1, acc[1][1]);
    wait_async();
    __syncthreads();
  }

  const int col  = lane & 15;
  const int row0 = (lane >> 4) << 3;
#pragma unroll
  for (int mf = 0; mf < 2; ++mf)
#pragma unroll
    for (int nf = 0; nf < 2; ++nf) {
      const int mb = mblock + wm * 32 + mf * 16;
      const int nb = nblock + wn * 32 + nf * 16;
#pragma unroll
      for (int r = 0; r < 8; ++r)
        out[(size_t)(mb + row0 + r) * TC + nb + col] = acc[mf][nf][r];
    }
}

// ---------------------------------------------------------------------------
// Launcher.  Workspace layout (bf16 elements):
//   x_bf @0 (4M) | wq_bf @4194304 | wk_bf @5242880 | wv_bf @6291456 |
//   wot @7340032 | q_bf @8388608 (4M) | k_bf @12582912 (4M) |
//   vt_bf @16777216 (4M) | y_bf @20971520 (4M)   -> 48 MiB total
// ---------------------------------------------------------------------------
extern "C" void kernel_launch(void* const* d_in, const int* in_sizes, int n_in,
                              void* d_out, int out_size, void* d_ws, size_t ws_size,
                              hipStream_t stream) {
  const float* x  = (const float*)d_in[0];
  const float* wq = (const float*)d_in[1];
  const float* wk = (const float*)d_in[2];
  const float* wv = (const float*)d_in[3];
  const float* wo = (const float*)d_in[4];
  float* out = (float*)d_out;

  bf16* ws    = (bf16*)d_ws;
  bf16* x_bf  = ws;
  bf16* wq_bf = ws + 4194304;
  bf16* wk_bf = ws + 5242880;
  bf16* wv_bf = ws + 6291456;
  bf16* wot   = ws + 7340032;
  bf16* q_bf  = ws + 8388608;
  bf16* k_bf  = ws + 12582912;
  bf16* vt_bf = ws + 16777216;
  bf16* y_bf  = ws + 20971520;

  cvt_f32_bf16<<<4096, 256, 0, stream>>>(x,  x_bf,  MROWS * TC);
  cvt_f32_bf16<<<1024, 256, 0, stream>>>(wq, wq_bf, TH * TD * TC);
  cvt_f32_bf16<<<1024, 256, 0, stream>>>(wk, wk_bf, TH * TD * TC);
  cvt_f32_bf16<<<1024, 256, 0, stream>>>(wv, wv_bf, TH * TD * TC);
  transpose_wo_bf16<<<4096, 256, 0, stream>>>(wo, wot);

  qkv_gemm<<<1536, 256, 0, stream>>>(x_bf, wq_bf, wk_bf, wv_bf, q_bf, k_bf, vt_bf);
  attn_fused<<<512, 256, 0, stream>>>(q_bf, k_bf, vt_bf, y_bf);
  out_gemm<<<512, 256, 0, stream>>>(y_bf, wot, out);
}
